// MultiHeadSelfAttention_2757369004352
// MI455X (gfx1250) — compile-verified, matrix operand order checked by reading, not run on hardware
//
#include <hip/hip_runtime.h>
#include <hip/hip_bf16.h>

typedef __attribute__((ext_vector_type(8)))  _Float16 v8h;
typedef __attribute__((ext_vector_type(16))) _Float16 v16h;
typedef __attribute__((ext_vector_type(8)))  float    v8f;

#define B_  4
#define S_  2048
#define D_  1024
#define H_  16
#define DH_ 64
#define NEGBIG 1e10f

union FragU { v16h v; v8h h[2]; };

// CDNA5 16-bit A/B fragment: per lane, element e holds k = (e>>3)*16 + half*8 + (e&7)
// => two contiguous 8-half (16B) loads at base+half*8 and base+16+half*8.
__device__ __forceinline__ v16h load_frag(const _Float16* base, int half) {
    FragU u;
    u.h[0] = *(const v8h*)(base + half * 8);
    u.h[1] = *(const v8h*)(base + 16 + half * 8);
    return u.v;
}

__device__ __forceinline__ v8f wmma_f16(v16h a, v16h b, v8f c) {
    return __builtin_amdgcn_wmma_f32_16x16x32_f16(false, a, false, b, (short)0, c, false, false);
}

// Async copy of one 16-byte chunk per lane: global -> LDS (tracked by ASYNCcnt).
__device__ __forceinline__ void async_b128(uint32_t lds_off, unsigned long long gaddr) {
    asm volatile("global_load_async_to_lds_b128 %0, %1, off"
                 :: "v"(lds_off), "v"(gaddr) : "memory");
}

// ---------------- fp32 -> f16 conversion (x row-major, W transposed) ----------------
__global__ void convert_kernel(const float* __restrict__ x,
                               const float* __restrict__ Wq,
                               const float* __restrict__ Wk,
                               const float* __restrict__ Wv,
                               _Float16* __restrict__ xh,
                               _Float16* __restrict__ wT) {
    const int NX = B_ * S_ * D_;      // 8388608
    const int NW = D_ * H_ * DH_;     // 1048576 (= 1<<20)
    const int total = NX + 3 * NW;
    for (int idx = blockIdx.x * blockDim.x + threadIdx.x; idx < total;
         idx += gridDim.x * blockDim.x) {
        if (idx < NX) {
            xh[idx] = (_Float16)x[idx];
        } else {
            int j = idx - NX;
            int m = j >> 20;              // which weight matrix
            int r = j & (NW - 1);
            int c = r >> 10;              // output column (row of WT)
            int k = r & 1023;             // reduction index
            const float* W = (m == 0) ? Wq : (m == 1) ? Wk : Wv;
            wT[(size_t)m * NW + r] = (_Float16)W[k * D_ + c];   // WT[c][k] = W[k][c]
        }
    }
}

// ---------------- QKV projection: one wave = one 16x64 tile (A reused 4x) ----------------
// grid = (8192/16, 3072/256) = (512, 12), block = 128 (4 waves)
__global__ void qkv_kernel(const _Float16* __restrict__ xh,
                           const _Float16* __restrict__ wT,
                           const float* __restrict__ bq,
                           const float* __restrict__ bk,
                           const float* __restrict__ bv,
                           _Float16* __restrict__ Qh,    // [B,H,S,DH]
                           _Float16* __restrict__ Kh,    // [B,H,S,DH]
                           _Float16* __restrict__ VhT) { // [B,H,DH,S]
    const int lane = threadIdx.x & 31;
    const int wave = threadIdx.x >> 5;
    const int n    = lane & 15;
    const int half = lane >> 4;
    const int row0  = blockIdx.x * 16;
    const int ctot0 = blockIdx.y * 256 + wave * 64;   // 64 output columns per wave
    const int mat   = ctot0 >> 10;                    // 0=Q, 1=K, 2=V
    const int c0    = ctot0 & 1023;

    const _Float16* arow = xh + (size_t)(row0 + n) * D_;
    const _Float16* brow = wT + (size_t)mat * D_ * D_ + (size_t)(c0 + n) * D_;

    v8f acc0 = {}, acc1 = {}, acc2 = {}, acc3 = {};
#pragma unroll 2
    for (int kk = 0; kk < D_; kk += 32) {
        v16h a  = load_frag(arow + kk, half);
        v16h f0 = load_frag(brow + kk,            half);
        v16h f1 = load_frag(brow + 16 * D_ + kk,  half);
        v16h f2 = load_frag(brow + 32 * D_ + kk,  half);
        v16h f3 = load_frag(brow + 48 * D_ + kk,  half);
        acc0 = wmma_f16(a, f0, acc0);
        acc1 = wmma_f16(a, f1, acc1);
        acc2 = wmma_f16(a, f2, acc2);
        acc3 = wmma_f16(a, f3, acc3);
    }

    const float* bias = (mat == 0) ? bq : (mat == 1) ? bk : bv;
#pragma unroll
    for (int g = 0; g < 4; ++g) {
        const int c  = c0 + g * 16 + n;
        const int hh = c >> 6, d = c & 63;
        const float bval = bias[c];
        const v8f acc = (g == 0) ? acc0 : (g == 1) ? acc1 : (g == 2) ? acc2 : acc3;
#pragma unroll
        for (int r = 0; r < 8; ++r) {
            int R  = row0 + half * 8 + r;
            int bb = R >> 11;                 // batch
            int s  = R & (S_ - 1);
            _Float16 val = (_Float16)(acc[r] + bval);
            if (mat == 0)
                Qh[(((size_t)(bb * H_ + hh)) * S_ + s) * DH_ + d] = val;
            else if (mat == 1)
                Kh[(((size_t)(bb * H_ + hh)) * S_ + s) * DH_ + d] = val;
            else
                VhT[(((size_t)(bb * H_ + hh)) * DH_ + d) * S_ + s] = val;
        }
    }
}

// ---- Flash attention: block-cooperative async K/V staging, 8 waves / 128 queries ----
// grid = (S/128, B*H) = (16, 64), block = 256 (8 waves)
__global__ void attn_kernel(const _Float16* __restrict__ Qh,
                            const _Float16* __restrict__ Kh,
                            const _Float16* __restrict__ VhT,
                            const float* __restrict__ mask,
                            float* __restrict__ out) {
    // block-shared double buffers: K tile [32 keys][64 dh], V tile [64 dh][32 keys]
    __shared__ _Float16 ldsK[2][32 * DH_];   // 2 x 4KB
    __shared__ _Float16 ldsV[2][DH_ * 32];   // 2 x 4KB
    __shared__ _Float16 ldsP[8][16 * 32];    // 8 x 1KB

    const int tid  = threadIdx.x;
    const int lane = tid & 31;
    const int wave = tid >> 5;
    const int n    = lane & 15;
    const int half = lane >> 4;
    const int bh = blockIdx.y;
    const int b  = bh >> 4;
    const int h  = bh & 15;
    const int q0 = blockIdx.x * 128 + wave * 16;

    const _Float16* Qb = Qh  + (size_t)bh * S_ * DH_;
    const _Float16* Kb = Kh  + (size_t)bh * S_ * DH_;
    const _Float16* Vb = VhT + (size_t)bh * DH_ * S_;
    const float*    mb = mask + (size_t)b * S_;

    // cooperative async stage of one 32-key chunk: 2 x b128 per thread (1 K, 1 V)
    auto stage_chunk = [&](int key0, int buf) {
        {
            int r = tid >> 3, c = tid & 7;      // K: 32 rows x 8 chunks of 16B
            async_b128((uint32_t)(uintptr_t)(&ldsK[buf][0] + r * DH_ + c * 8),
                       (unsigned long long)(uintptr_t)(Kb + (size_t)(key0 + r) * DH_ + c * 8));
        }
        {
            int r = tid >> 2, c = tid & 3;      // V: 64 rows x 4 chunks of 16B
            async_b128((uint32_t)(uintptr_t)(&ldsV[buf][0] + r * 32 + c * 8),
                       (unsigned long long)(uintptr_t)(Vb + (size_t)r * S_ + key0 + c * 8));
        }
    };

    // Q A-fragments, two k-chunks covering DH=64; resident for the whole loop
    v16h qa0, qa1;
    {
        const _Float16* qrow = Qb + (size_t)(q0 + n) * DH_;
        qa0 = load_frag(qrow, half);
        qa1 = load_frag(qrow + 32, half);
    }

    v8f o0 = {}, o1 = {}, o2 = {}, o3 = {};
    float mrow[8], lrow[8];
#pragma unroll
    for (int r = 0; r < 8; ++r) { mrow[r] = -3.0e38f; lrow[r] = 0.0f; }
    const float scale = 0.125f;  // 1/sqrt(DH)

    // prologue: chunk 0 -> buffer 0; own portion done, then block-wide visibility
    stage_chunk(0, 0);
    asm volatile("s_wait_asynccnt 0x0" ::: "memory");
    __syncthreads();

    for (int key0 = 0; key0 < S_; key0 += 32) {
        const int buf = (key0 >> 5) & 1;
        const bool have_next = (key0 + 32) < S_;
        if (have_next) {
            // insurance: this wave's DS reads of buf^1 (iter-1) are complete
            asm volatile("s_wait_dscnt 0x0" ::: "memory");
            stage_chunk(key0 + 32, buf ^ 1);   // overlaps with compute below
        }

        const _Float16* bK = &ldsK[buf][0];
        const _Float16* bV = &ldsV[buf][0];

        // ---- scores: two 16-key tiles from LDS, K-chunks over DH ----
        v8f sc0 = {}, sc1 = {};
        {
            const _Float16* krow0 = bK + (size_t)n * DH_;
            const _Float16* krow1 = bK + (size_t)(16 + n) * DH_;
            v16h b00 = load_frag(krow0,      half);
            v16h b01 = load_frag(krow0 + 32, half);
            v16h b10 = load_frag(krow1,      half);
            v16h b11 = load_frag(krow1 + 32, half);
            sc0 = wmma_f16(qa0, b00, sc0);
            sc0 = wmma_f16(qa1, b01, sc0);
            sc1 = wmma_f16(qa0, b10, sc1);
            sc1 = wmma_f16(qa1, b11, sc1);
        }

        const float mv0 = mb[key0 + n];
        const float mv1 = mb[key0 + 16 + n];
        const float pen0 = (1.0f - mv0) * NEGBIG;
        const float pen1 = (1.0f - mv1) * NEGBIG;

        float p0[8], p1[8], alpha[8];
#pragma unroll
        for (int r = 0; r < 8; ++r) {
            float s0 = sc0[r] * scale - pen0;
            float s1 = sc1[r] * scale - pen1;
            float rmax = fmaxf(s0, s1);
#pragma unroll
            for (int off = 1; off < 16; off <<= 1)
                rmax = fmaxf(rmax, __shfl_xor(rmax, off, 32));
            float mnew = fmaxf(mrow[r], rmax);
            alpha[r] = __expf(mrow[r] - mnew);
            float e0 = __expf(s0 - mnew) * mv0;   // probs * mask, as in reference
            float e1 = __expf(s1 - mnew) * mv1;
            float rsum = e0 + e1;
#pragma unroll
            for (int off = 1; off < 16; off <<= 1)
                rsum += __shfl_xor(rsum, off, 32);
            lrow[r] = lrow[r] * alpha[r] + rsum;
            mrow[r] = mnew;
            p0[r] = e0; p1[r] = e1;
        }

        // ---- rescale O, stage P (C-layout -> A-layout) via per-wave LDS ----
        _Float16* pl = &ldsP[wave][0];
#pragma unroll
        for (int r = 0; r < 8; ++r) {
            o0[r] *= alpha[r]; o1[r] *= alpha[r];
            o2[r] *= alpha[r]; o3[r] *= alpha[r];
            int m = half * 8 + r;
            pl[m * 32 + n]      = (_Float16)p0[r];
            pl[m * 32 + 16 + n] = (_Float16)p1[r];
        }
        v16h pa;
        {
            FragU u;
            const _Float16* prow = pl + n * 32;
            u.h[0] = *(const v8h*)(prow + half * 8);
            u.h[1] = *(const v8h*)(prow + 16 + half * 8);
            pa = u.v;
        }

        // ---- O += P(16x32) x V(32x64), V B-fragments from LDS ----
        {
            const _Float16* vcol = bV + (size_t)n * 32;
            v16h vb0 = load_frag(vcol,            half);
            v16h vb1 = load_frag(vcol + 16 * 32,  half);
            v16h vb2 = load_frag(vcol + 32 * 32,  half);
            v16h vb3 = load_frag(vcol + 48 * 32,  half);
            o0 = wmma_f16(pa, vb0, o0);
            o1 = wmma_f16(pa, vb1, o1);
            o2 = wmma_f16(pa, vb2, o2);
            o3 = wmma_f16(pa, vb3, o3);
        }

        // own async portion of the next chunk is done -> make visible block-wide
        asm volatile("s_wait_asynccnt 0x0" ::: "memory");
        __syncthreads();
    }

    // ---- finalize: O / l, store fp32 output [B, S, H*DH] ----
#pragma unroll
    for (int r = 0; r < 8; ++r) {
        float inv = 1.0f / (lrow[r] + 1e-30f);
        int s = q0 + half * 8 + r;
        float* orow = out + ((size_t)(b * S_ + s)) * (H_ * DH_) + h * DH_;
        orow[ 0 + n] = o0[r] * inv;
        orow[16 + n] = o1[r] * inv;
        orow[32 + n] = o2[r] * inv;
        orow[48 + n] = o3[r] * inv;
    }
}

extern "C" void kernel_launch(void* const* d_in, const int* in_sizes, int n_in,
                              void* d_out, int out_size, void* d_ws, size_t ws_size,
                              hipStream_t stream) {
    const float* x    = (const float*)d_in[0];
    const float* mask = (const float*)d_in[1];
    const float* Wq   = (const float*)d_in[2];
    const float* bq   = (const float*)d_in[3];
    const float* Wk   = (const float*)d_in[4];
    const float* bk   = (const float*)d_in[5];
    const float* Wv   = (const float*)d_in[6];
    const float* bv   = (const float*)d_in[7];
    float* out = (float*)d_out;

    const size_t NX = (size_t)B_ * S_ * D_;        // 8388608
    const size_t NW = (size_t)D_ * H_ * DH_;       // 1048576
    const size_t NQ = (size_t)B_ * H_ * S_ * DH_;  // 8388608

    _Float16* xh  = (_Float16*)d_ws;
    _Float16* wT  = xh + NX;
    _Float16* Qh  = wT + 3 * NW;
    _Float16* Kh  = Qh + NQ;
    _Float16* VhT = Kh + NQ;

    {
        const int total = (int)(NX + 3 * NW);
        int blocks = (total + 255) / 256;
        convert_kernel<<<blocks, 256, 0, stream>>>(x, Wq, Wk, Wv, xh, wT);
    }
    {
        dim3 grid((B_ * S_) / 16, (3 * H_ * DH_) / 256);  // (512, 12)
        qkv_kernel<<<grid, 128, 0, stream>>>(xh, wT, bq, bk, bv, Qh, Kh, VhT);
    }
    {
        dim3 grid(S_ / 128, B_ * H_);                      // (16, 64)
        attn_kernel<<<grid, 256, 0, stream>>>(Qh, Kh, VhT, mask, out);
    }
}
